// selfattention_8967891714360
// MI455X (gfx1250) — matrix operand order repaired and checked by reading, not compiled, not run on hardware
//
#include <hip/hip_runtime.h>
#include <hip/hip_bf16.h>

typedef __attribute__((ext_vector_type(16))) __bf16       v16bf;
typedef __attribute__((ext_vector_type(8)))  float        v8f;
typedef __attribute__((ext_vector_type(8)))  unsigned int v8u;
typedef __attribute__((__vector_size__(4 * sizeof(int)))) int v4i;

#define B_DIM  4
#define C_DIM  256
#define CK_DIM 32
#define N_DIM  4096
#define O_DIM  (C_DIM + 2 * CK_DIM)   // 320 projection outputs
#define VPITCH 40                      // LDS row pitch in bf16 (80B, 16B aligned)
#define NCHUNK (N_DIM / 32)

#define AS1 __attribute__((address_space(1)))
#define AS3 __attribute__((address_space(3)))

#if defined(__gfx1250__) &&                                            \
    __has_builtin(__builtin_amdgcn_global_load_async_to_lds_b128) &&   \
    __has_builtin(__builtin_amdgcn_s_wait_asynccnt)
#define USE_ASYNC_LDS 1
#else
#define USE_ASYNC_LDS 0
#endif

#if __has_builtin(__builtin_amdgcn_sched_barrier)
#define SCHED_FENCE() __builtin_amdgcn_sched_barrier(0)
#else
#define SCHED_FENCE()
#endif

__device__ __forceinline__ unsigned rnd_bf16(unsigned u) {
  return (u + 0x7FFFu + ((u >> 16) & 1u)) >> 16;   // RNE f32 -> bf16
}
__device__ __forceinline__ unsigned pack2_bf16(float lo, float hi) {
  unsigned a = rnd_bf16(__float_as_uint(lo));
  unsigned b = rnd_bf16(__float_as_uint(hi));
  return a | (b << 16);
}
__device__ __forceinline__ int apair(int vv, int h) {  // A-operand dword offset
  return vv + 4 * h + (vv < 4 ? 0 : 4);
}

// 16B global -> LDS copy: async DMA on CDNA5, sync fallback otherwise.
__device__ __forceinline__ void copy16(const unsigned short* gsrc,
                                       unsigned short* ldst) {
#if USE_ASYNC_LDS
  __builtin_amdgcn_global_load_async_to_lds_b128(
      (AS1 v4i*)(AS1 void*)(void*)gsrc, (AS3 v4i*)(AS3 void*)ldst, 0, 0);
#else
  *(uint4*)ldst = *(const uint4*)gsrc;
#endif
}
__device__ __forceinline__ void async_fence() {
#if USE_ASYNC_LDS
  __builtin_amdgcn_s_wait_asynccnt(0);
#endif
}

// ---------------------------------------------------------------------------
// Kernel P0: convert concatenated projection weights [wq;wk;wv] to bf16
// wb[o][c], o in 0..319.  grid = 320, block = 256
// ---------------------------------------------------------------------------
__global__ void wconv_kernel(const float* __restrict__ wq,
                             const float* __restrict__ wk,
                             const float* __restrict__ wv,
                             unsigned short* __restrict__ wb) {
  const int o = blockIdx.x;
  const int c = threadIdx.x;
  float w;
  if (o < CK_DIM)          w = wq[(size_t)o * C_DIM + c];
  else if (o < 2 * CK_DIM) w = wk[(size_t)(o - CK_DIM) * C_DIM + c];
  else                     w = wv[(size_t)(o - 2 * CK_DIM) * C_DIM + c];
  wb[(size_t)o * C_DIM + c] = (unsigned short)rnd_bf16(__float_as_uint(w));
}

// ---------------------------------------------------------------------------
// Kernel P1: WMMA projection.  One wave per 16-position tile computes all
// 320 outputs:  S[o,i] = sum_c W[o,c] * X[c,i] + bias.
// Weight A-operands are software-pipelined (load t+1 while WMMA t runs);
// a sched_barrier keeps the backend from sinking the prefetch.
// q,k -> bf16 [B,N,CK]; v -> bf16 [B,C,N].  grid = B*N/16, block = 32
// ---------------------------------------------------------------------------
__global__ void __launch_bounds__(32)
proj_kernel(const float* __restrict__ x,
            const unsigned short* __restrict__ wb,
            const float* __restrict__ bq, const float* __restrict__ bk,
            const float* __restrict__ bv,
            unsigned short* __restrict__ q,
            unsigned short* __restrict__ k,
            unsigned short* __restrict__ v) {
  const int ntile = N_DIM / 16;
  const int b     = blockIdx.x / ntile;
  const int itile = (blockIdx.x % ntile) * 16;
  const int lane  = threadIdx.x;
  const int h     = lane >> 4;
  const int ln    = lane & 15;
  const int i     = itile + ln;

  const unsigned* wbdw = (const unsigned*)wb;

  v8f acc[O_DIM / 16];
#pragma unroll
  for (int t = 0; t < O_DIM / 16; ++t)
#pragma unroll
    for (int r = 0; r < 8; ++r) acc[t][r] = 0.0f;

  auto load_wa = [&](int t, int c0) {
    const int o = t * 16 + ln;
    const int rowbase = o * (C_DIM / 2) + c0 / 2;
    v8u wa;
#pragma unroll
    for (int vv = 0; vv < 8; ++vv) wa[vv] = wbdw[rowbase + apair(vv, h)];
    return wa;
  };

#pragma unroll 1
  for (int cc = 0; cc < C_DIM / 32; ++cc) {
    const int c0 = cc * 32;
    // B operand: X chunk (32c x 16i); lane = i, vgpr vv: c = c0 + 2*apair(vv,h)
    v8u xb;
#pragma unroll
    for (int vv = 0; vv < 8; ++vv) {
      const int c = c0 + 2 * apair(vv, h);
      const float lo = x[((size_t)(b * C_DIM + c))     * N_DIM + i];
      const float hi = x[((size_t)(b * C_DIM + c + 1)) * N_DIM + i];
      xb[vv] = pack2_bf16(lo, hi);
    }
    const v16bf xB = __builtin_bit_cast(v16bf, xb);

    v8u wa_cur = load_wa(0, c0);
#pragma unroll
    for (int t = 0; t < O_DIM / 16; ++t) {
      v8u wa_next = (t + 1 < O_DIM / 16) ? load_wa(t + 1, c0) : wa_cur;
      SCHED_FENCE();   // keep next-tile loads in flight across this WMMA
      acc[t] = __builtin_amdgcn_wmma_f32_16x16x32_bf16(
          false, __builtin_bit_cast(v16bf, wa_cur), false, xB, (short)0,
          acc[t], false, false);
      SCHED_FENCE();
      wa_cur = wa_next;
    }
  }

  // epilogue: add bias, convert, store
#pragma unroll
  for (int t = 0; t < O_DIM / 16; ++t) {
    float val[8];
#pragma unroll
    for (int r = 0; r < 8; ++r) {
      const int o = t * 16 + r + 8 * h;
      float bias;
      if (o < CK_DIM)          bias = bq[o];
      else if (o < 2 * CK_DIM) bias = bk[o - CK_DIM];
      else                     bias = bv[o - 2 * CK_DIM];
      val[r] = acc[t][r] + bias;
    }
    if (t < 4) {  // q (t=0,1) / k (t=2,3): 8 consecutive bf16 -> one b128
      uint4 st;
      st.x = pack2_bf16(val[0], val[1]);
      st.y = pack2_bf16(val[2], val[3]);
      st.z = pack2_bf16(val[4], val[5]);
      st.w = pack2_bf16(val[6], val[7]);
      unsigned short* dst = (t < 2 ? q : k);
      const size_t ofs = ((size_t)(b * N_DIM + i)) * CK_DIM + (t & 1) * 16 + 8 * h;
      *(uint4*)(dst + ofs) = st;
    } else {      // v: rows of [C, N], stride N per r
#pragma unroll
      for (int r = 0; r < 8; ++r) {
        const int o = t * 16 + r + 8 * h - 2 * CK_DIM;
        v[((size_t)(b * C_DIM + o)) * N_DIM + i] =
            (unsigned short)rnd_bf16(__float_as_uint(val[r]));
      }
    }
  }
}

// ---------------------------------------------------------------------------
// Stage one 32-key chunk of K (32x32 bf16) and V (256x32 bf16) into LDS.
// 128 threads; 9 x 16B transfers per thread (async DMA on CDNA5).
// ---------------------------------------------------------------------------
__device__ __forceinline__ void stage_chunk(int b, int jc, int tid,
                                            const unsigned* kdw,
                                            const unsigned* vdw,
                                            unsigned short* lk,
                                            unsigned short* lv) {
  const int j0 = jc * 32;
  const int jr = tid >> 2, qt = tid & 3;  // K: 32 rows x 4 x b128
  const unsigned short* ksrc =
      (const unsigned short*)(kdw + ((size_t)(b * N_DIM) + j0 + jr) * (CK_DIM / 2)) +
      qt * 8;
  copy16(ksrc, lk + jr * VPITCH + qt * 8);
#pragma unroll
  for (int rr = 0; rr < 2; ++rr) {        // V: 2 rows x 4 x b128 per thread
    const int c = rr * 128 + tid;
    const unsigned short* vsrc =
        (const unsigned short*)(vdw + ((size_t)(b * C_DIM + c)) * (N_DIM / 2) +
                                jc * 16);
    unsigned short* vdst = lv + c * VPITCH;
#pragma unroll
    for (int p = 0; p < 4; ++p) copy16(vsrc + p * 8, vdst + p * 8);
  }
}

// ---------------------------------------------------------------------------
// Kernel 2: fused flash attention + gamma*out + x.
// 4 waves / WG, each owning a 16-query tile; K/V chunks double-buffered in
// LDS via async global->LDS DMA.  S^T = K@Q^T keeps softmax rows per-lane;
// O^T = V@P^T lands in [c,i] output orientation.  grid = B*N/64, block = 128
// ---------------------------------------------------------------------------
__global__ void __launch_bounds__(128)
attn_kernel(const unsigned short* __restrict__ qg,
            const unsigned short* __restrict__ kg,
            const unsigned short* __restrict__ vg,
            const float* __restrict__ x,
            const float* __restrict__ gamma_p,
            float* __restrict__ out) {
  const int nwg   = N_DIM / 64;
  const int b     = blockIdx.x / nwg;
  const int tid   = threadIdx.x;
  const int w     = tid >> 5;
  const int itile = (blockIdx.x % nwg) * 64 + w * 16;
  const int lane  = tid & 31;
  const int h     = lane >> 4;
  const int ln    = lane & 15;

  __shared__ __align__(16) unsigned short lsK[2][32 * VPITCH];
  __shared__ __align__(16) unsigned short lsV[2][C_DIM * VPITCH];

  const unsigned* qdw = (const unsigned*)qg;
  const unsigned* kdw = (const unsigned*)kg;
  const unsigned* vdw = (const unsigned*)vg;

  // Q^T B-operand (32ck x 16i): lane = i; vgpr vv: ck = 2*apair(vv,h)
  v8u qb;
  {
    const int rowbase = ((b * N_DIM) + itile + ln) * (CK_DIM / 2);
#pragma unroll
    for (int vv = 0; vv < 8; ++vv) qb[vv] = qdw[rowbase + apair(vv, h)];
  }
  const v16bf qB = __builtin_bit_cast(v16bf, qb);

  v8f acc[16];
#pragma unroll
  for (int t = 0; t < 16; ++t)
#pragma unroll
    for (int r = 0; r < 8; ++r) acc[t][r] = 0.0f;

  float m = -3.0e38f, l = 0.0f;
  v8f zero;
#pragma unroll
  for (int r = 0; r < 8; ++r) zero[r] = 0.0f;

  // prologue: fill buffer 0
  stage_chunk(b, 0, tid, kdw, vdw, lsK[0], lsV[0]);
  async_fence();
  __syncthreads();

#pragma unroll 1
  for (int jc = 0; jc < NCHUNK; ++jc) {
    const int cur = jc & 1;
    if (jc + 1 < NCHUNK)   // async-fill the other buffer while computing
      stage_chunk(b, jc + 1, tid, kdw, vdw, lsK[cur ^ 1], lsV[cur ^ 1]);

    const unsigned short* lk = lsK[cur];
    const unsigned short* lv = lsV[cur];

    // ---- S^T tiles: K A-operands read from LDS ----
    v8u ka0, ka1;
    {
      const uint4* kr0 = (const uint4*)(lk + ln * VPITCH);
      const uint4* kr1 = (const uint4*)(lk + (16 + ln) * VPITCH);
      const uint4 a0 = kr0[h], a1 = kr0[2 + h];
      const uint4 b0 = kr1[h], b1 = kr1[2 + h];
      ka0[0] = a0.x; ka0[1] = a0.y; ka0[2] = a0.z; ka0[3] = a0.w;
      ka0[4] = a1.x; ka0[5] = a1.y; ka0[6] = a1.z; ka0[7] = a1.w;
      ka1[0] = b0.x; ka1[1] = b0.y; ka1[2] = b0.z; ka1[3] = b0.w;
      ka1[4] = b1.x; ka1[5] = b1.y; ka1[6] = b1.z; ka1[7] = b1.w;
    }

    v8f st0 = __builtin_amdgcn_wmma_f32_16x16x32_bf16(
        false, __builtin_bit_cast(v16bf, ka0), false, qB, (short)0, zero, false, false);
    v8f st1 = __builtin_amdgcn_wmma_f32_16x16x32_bf16(
        false, __builtin_bit_cast(v16bf, ka1), false, qB, (short)0, zero, false, false);

    // ---- online softmax (row = query i = lane column, xor-16 pair) ----
    float rm = -3.0e38f;
#pragma unroll
    for (int r = 0; r < 8; ++r) { rm = fmaxf(rm, st0[r]); rm = fmaxf(rm, st1[r]); }
    rm = fmaxf(rm, __shfl_xor(rm, 16, 32));
    const float m_new = fmaxf(m, rm);
    const float alpha = __expf(m - m_new);

    float rs = 0.0f;
#pragma unroll
    for (int r = 0; r < 8; ++r) {
      st0[r] = __expf(st0[r] - m_new);
      st1[r] = __expf(st1[r] - m_new);
      rs += st0[r] + st1[r];
    }
    rs += __shfl_xor(rs, 16, 32);
    l = l * alpha + rs;
    m = m_new;

#pragma unroll
    for (int t = 0; t < 16; ++t)
#pragma unroll
      for (int r = 0; r < 8; ++r) acc[t][r] *= alpha;

    // ---- P^T (D layout) -> B operand (32j x 16i) via xor-16 shuffles ----
    v8u pb;
#pragma unroll
    for (int vv = 0; vv < 4; ++vv) {
      const float lo0 = st0[2 * vv],     hi0 = st0[2 * vv + 1];
      const float lo1 = __shfl_xor(st1[2 * vv], 16, 32);
      const float hi1 = __shfl_xor(st1[2 * vv + 1], 16, 32);
      pb[vv] = pack2_bf16(h ? lo1 : lo0, h ? hi1 : hi0);
    }
#pragma unroll
    for (int vv = 4; vv < 8; ++vv) {
      const int r2 = 2 * (vv - 4);
      const float lo0 = __shfl_xor(st0[r2], 16, 32);
      const float hi0 = __shfl_xor(st0[r2 + 1], 16, 32);
      const float lo1 = st1[r2], hi1 = st1[r2 + 1];
      pb[vv] = pack2_bf16(h ? lo1 : lo0, h ? hi1 : hi0);
    }
    const v16bf pB = __builtin_bit_cast(v16bf, pb);

    // ---- O^T += V_tile @ P^T over all 16 channel tiles (V from LDS) ----
#pragma unroll
    for (int t = 0; t < 16; ++t) {
      const uint4* vr = (const uint4*)(lv + (t * 16 + ln) * VPITCH);
      const uint4 v0 = vr[2 * h], v1 = vr[2 * h + 1];
      v8u va;
      va[0] = v0.x; va[1] = v0.y; va[2] = v0.z; va[3] = v0.w;
      va[4] = v1.x; va[5] = v1.y; va[6] = v1.z; va[7] = v1.w;
      acc[t] = __builtin_amdgcn_wmma_f32_16x16x32_bf16(
          false, __builtin_bit_cast(v16bf, va), false, pB, (short)0, acc[t],
          false, false);
    }

    async_fence();        // our async fills must land before the barrier
    __syncthreads();      // everyone done reading cur / writing next
  }

  // ---- epilogue: out = gamma * (O / l) + x ----
  const float inv_l = 1.0f / l;
  const float gamma = gamma_p[0];
#pragma unroll
  for (int t = 0; t < 16; ++t) {
#pragma unroll
    for (int r = 0; r < 8; ++r) {
      const int c = t * 16 + r + 8 * h;
      const size_t idx = ((size_t)(b * C_DIM + c)) * N_DIM + itile + ln;
      out[idx] = gamma * (acc[t][r] * inv_l) + x[idx];
    }
  }
}

// ---------------------------------------------------------------------------
extern "C" void kernel_launch(void* const* d_in, const int* in_sizes, int n_in,
                              void* d_out, int out_size, void* d_ws, size_t ws_size,
                              hipStream_t stream) {
  const float* x     = (const float*)d_in[0];
  const float* wq    = (const float*)d_in[1];
  const float* bq    = (const float*)d_in[2];
  const float* wk    = (const float*)d_in[3];
  const float* bk    = (const float*)d_in[4];
  const float* wv    = (const float*)d_in[5];
  const float* bv    = (const float*)d_in[6];
  const float* gamma = (const float*)d_in[7];

  unsigned short* ws = (unsigned short*)d_ws;
  unsigned short* dq = ws;                                           // 1 MB
  unsigned short* dk = dq + (size_t)B_DIM * N_DIM * CK_DIM;          // 1 MB
  unsigned short* dv = dk + (size_t)B_DIM * N_DIM * CK_DIM;          // 8 MB
  unsigned short* wb = dv + (size_t)B_DIM * C_DIM * N_DIM;           // 160 KB

  wconv_kernel<<<dim3(O_DIM), dim3(C_DIM), 0, stream>>>(wq, wk, wv, wb);

  proj_kernel<<<dim3(B_DIM * N_DIM / 16), dim3(32), 0, stream>>>(
      x, wb, bq, bk, bv, dq, dk, dv);

  attn_kernel<<<dim3(B_DIM * N_DIM / 64), dim3(128), 0, stream>>>(
      dq, dk, dv, x, gamma, (float*)d_out);
}
